// MultiHeadDotProductAttention_18769007083643
// MI455X (gfx1250) — matrix operand order repaired and checked by reading, not compiled, not run on hardware
//
#include <hip/hip_runtime.h>
#include <hip/hip_bf16.h>

typedef __bf16 bf16_t;
typedef __attribute__((ext_vector_type(16))) __bf16 v16bf;
typedef __attribute__((ext_vector_type(8)))  __bf16 v8bf;
typedef __attribute__((ext_vector_type(4)))  __bf16 v4bf;
typedef __attribute__((ext_vector_type(8)))  float  v8f;

#define EDGES   5
#define BATCH   4
#define NNODES  1024
#define DMODEL  768
#define NHEADS  12
#define DK      64
#define MTOT    (BATCH * NNODES)   // 4096

// CDNA5 async global->LDS path (ASYNCcnt-tracked, bypasses VGPRs), guarded so
// the file still compiles on toolchains without the builtins.
#if defined(__has_builtin)
#if __has_builtin(__builtin_amdgcn_global_load_async_to_lds_b128) && \
    __has_builtin(__builtin_amdgcn_s_wait_asynccnt)
#define USE_ASYNC_LDS 1
#endif
#endif

// ---------------------------------------------------------------------------
// Fragment builders (contiguous k = 16*(lane/16) + s for A and B alike, wave32)
// ---------------------------------------------------------------------------

// A-matrix 16x32 bf16: lane 0-15 -> M=lane, K=0..15;
// lane 16-31 -> M=lane-16, K=16..31. One contiguous 32B load.
__device__ __forceinline__ v16bf load_a_frag(const bf16_t* __restrict__ base,
                                             int ld, int lane) {
    int row = lane & 15;
    int kb  = (lane >> 4) << 4;          // 0 or 16
    const bf16_t* p = base + row * ld;
    return *(const v16bf*)(p + kb);
}

// B-matrix 32x16 bf16 where B = R^T of a row-major R[n][k] (ld = row stride):
// lane 0-15 -> N=lane, K=0..15; lane 16-31 -> N=lane-16, K=16..31.
// One contiguous 32B load per lane.
__device__ __forceinline__ v16bf load_bT_frag(const bf16_t* __restrict__ base,
                                              int ld, int lane) {
    int col = lane & 15;
    int kb  = (lane >> 4) << 4;          // 0 or 16
    return *(const v16bf*)(base + col * ld + kb);
}

// ---------------------------------------------------------------------------
// 1) fp32 -> bf16 conversion (float4 -> 4x bf16 per thread)
// ---------------------------------------------------------------------------
__global__ __launch_bounds__(256) void cvt_f32_bf16(const float* __restrict__ src,
                                                    bf16_t* __restrict__ dst,
                                                    int n4) {
    int i = blockIdx.x * 256 + threadIdx.x;
    if (i < n4) {
        float4 f = ((const float4*)src)[i];
        v4bf o;
        o[0] = (bf16_t)f.x; o[1] = (bf16_t)f.y;
        o[2] = (bf16_t)f.z; o[3] = (bf16_t)f.w;
        ((v4bf*)dst)[i] = o;
    }
}

// ---------------------------------------------------------------------------
// 2) Q/K projection GEMMs: out[e] = h @ W[e]^T + b[e]  (bf16 in, f32 acc, bf16 out)
//    grid = (MTOT/16, DMODEL/128, 2*EDGES); block = 256 (8 waves)
//    blockIdx.z: 0..4 -> Q edge z ; 5..9 -> K edge z-5
// ---------------------------------------------------------------------------
__global__ __launch_bounds__(256) void gemm_qk_kernel(
    const bf16_t* __restrict__ hbf,
    const bf16_t* __restrict__ wqbf, const bf16_t* __restrict__ wkbf,
    const float*  __restrict__ bq,   const float*  __restrict__ bk,
    bf16_t* __restrict__ qbf, bf16_t* __restrict__ kbf) {

    __shared__ bf16_t hsm[16 * DMODEL];           // 24 KB row-major h tile

    const int m0 = blockIdx.x * 16;
    const int n0 = blockIdx.y * 128;
    const int z  = blockIdx.z;
    const int e  = (z >= EDGES) ? (z - EDGES) : z;
    const bool is_k = (z >= EDGES);

    const bf16_t* w   = (is_k ? wkbf : wqbf) + (size_t)e * DMODEL * DMODEL;
    const float*  bia = (is_k ? bk   : bq)   + e * DMODEL;
    bf16_t*       out = (is_k ? kbf  : qbf)  + (size_t)e * MTOT * DMODEL;

    // stage 16 x 768 bf16 tile of h into LDS
    {
#ifdef USE_ASYNC_LDS
        // 24576 B = 1536 x b128 async transfers, 6 per thread, no VGPR staging
        typedef int v4i_t __attribute__((vector_size(16)));
        typedef __attribute__((address_space(3))) v4i_t* lds_v4i_p;
        char* s = (char*)(hbf + (size_t)m0 * DMODEL);
        char* d = (char*)hsm;
#pragma unroll
        for (int i = 0; i < 6; ++i) {
            int off = ((int)threadIdx.x + i * 256) * 16;
            __builtin_amdgcn_global_load_async_to_lds_b128(
                (v4i_t*)(s + off), (lds_v4i_p)(d + off), 0, 0);
        }
        __builtin_amdgcn_s_wait_asynccnt(0);
#else
        const uint32_t* s = (const uint32_t*)(hbf + (size_t)m0 * DMODEL);
        uint32_t* d = (uint32_t*)hsm;
        for (int i = threadIdx.x; i < 16 * (DMODEL / 2); i += 256) d[i] = s[i];
#endif
    }
    __syncthreads();

    const int wave = threadIdx.x >> 5;
    const int lane = threadIdx.x & 31;
    const int nc   = n0 + wave * 16;              // this wave's 16 output cols

    v8f acc = {};
#pragma unroll 4
    for (int kt = 0; kt < DMODEL; kt += 32) {
        v16bf a = load_a_frag(hsm + kt, DMODEL, lane);
        v16bf b = load_bT_frag(w + (size_t)nc * DMODEL + kt, DMODEL, lane);
        acc = __builtin_amdgcn_wmma_f32_16x16x32_bf16(false, a, false, b,
                                                      (short)0, acc, false, false);
    }

    const int col = nc + (lane & 15);
    const int rb  = (lane >> 4) << 3;
    const float bv = bia[col];
#pragma unroll
    for (int r = 0; r < 8; ++r) {
        int row = m0 + rb + r;
        out[(size_t)row * DMODEL + col] = (bf16_t)(acc[r] + bv);
    }
}

// ---------------------------------------------------------------------------
// 3) Edge-selected scores: out[h][b][n][m] = adj-select_e( Q_e K_e^T * scale ),
//    zeros -> -1e9.  grid = (N/16, NHEADS, BATCH); block = 256 (8 waves).
//    Each wave keeps all 10 Q A-fragments (5 edges x 2 k-tiles of 32) resident
//    and marches over column tiles; next tile's K slice is prefetched.
// ---------------------------------------------------------------------------
__global__ __launch_bounds__(256) void score_kernel(
    const bf16_t* __restrict__ qbf, const bf16_t* __restrict__ kbf,
    const int* __restrict__ adj, float* __restrict__ out) {

    const int row0 = blockIdx.x * 16;
    const int h    = blockIdx.y;
    const int b    = blockIdx.z;
    const int wave = threadIdx.x >> 5;
    const int lane = threadIdx.x & 31;

    v16bf aq[EDGES][2];
#pragma unroll
    for (int e = 0; e < EDGES; ++e) {
        const bf16_t* qb = qbf + ((size_t)(e * BATCH + b) * NNODES + row0) * DMODEL + h * DK;
        aq[e][0] = load_a_frag(qb,      DMODEL, lane);
        aq[e][1] = load_a_frag(qb + 32, DMODEL, lane);
    }

    const int rb = (lane >> 4) << 3;
    const int cl = lane & 15;

    for (int ct = wave; ct < NNODES / 16; ct += 8) {
        const int col0 = ct * 16;

        // prefetch next column tile's K slice (e=0 stream primes the L2 path)
        if (ct + 8 < NNODES / 16) {
            const bf16_t* kp = kbf + ((size_t)b * NNODES + col0 + 128 + cl) * DMODEL + h * DK;
            __builtin_prefetch(kp, 0, 1);
        }

        v8f acc[EDGES];
#pragma unroll
        for (int e = 0; e < EDGES; ++e) {
            const bf16_t* kb = kbf + ((size_t)(e * BATCH + b) * NNODES + col0) * DMODEL + h * DK;
            v8f c = {};
            v16bf b0 = load_bT_frag(kb,      DMODEL, lane);
            c = __builtin_amdgcn_wmma_f32_16x16x32_bf16(false, aq[e][0], false, b0,
                                                        (short)0, c, false, false);
            v16bf b1 = load_bT_frag(kb + 32, DMODEL, lane);
            c = __builtin_amdgcn_wmma_f32_16x16x32_bf16(false, aq[e][1], false, b1,
                                                        (short)0, c, false, false);
            acc[e] = c;
        }

        const int* adjp = adj + ((size_t)b * NNODES + row0 + rb) * NNODES + col0 + cl;
        float*     op   = out + ((size_t)(h * BATCH + b) * NNODES + row0 + rb) * NNODES + col0 + cl;
#pragma unroll
        for (int r = 0; r < 8; ++r) {
            int av = adjp[(size_t)r * NNODES];
            float v = 0.0f;
#pragma unroll
            for (int e = 0; e < EDGES; ++e)
                v = (av == e + 1) ? acc[e][r] * 0.125f : v;   // scale = 1/sqrt(64)
            if (v == 0.0f) v = -1.0e9f;
            op[(size_t)r * NNODES] = v;
        }
    }
}

// ---------------------------------------------------------------------------
// 4) Row softmax in place over d_out rows of 1024. One block per row.
// ---------------------------------------------------------------------------
__global__ __launch_bounds__(256) void softmax_kernel(float* __restrict__ out) {
    __shared__ float red[256];
    float* p = out + (size_t)blockIdx.x * NNODES;
    const int t = threadIdx.x;

    float v[4];
#pragma unroll
    for (int i = 0; i < 4; ++i) v[i] = p[t + 256 * i];

    float m = fmaxf(fmaxf(v[0], v[1]), fmaxf(v[2], v[3]));
    red[t] = m; __syncthreads();
    for (int s = 128; s > 0; s >>= 1) {
        if (t < s) red[t] = fmaxf(red[t], red[t + s]);
        __syncthreads();
    }
    m = red[0]; __syncthreads();

    float sum = 0.0f;
#pragma unroll
    for (int i = 0; i < 4; ++i) { v[i] = __expf(v[i] - m); sum += v[i]; }
    red[t] = sum; __syncthreads();
    for (int s = 128; s > 0; s >>= 1) {
        if (t < s) red[t] += red[t + s];
        __syncthreads();
    }
    const float inv = 1.0f / red[0];
#pragma unroll
    for (int i = 0; i < 4; ++i) p[t + 256 * i] = v[i] * inv;
}

// ---------------------------------------------------------------------------
// launch
// ---------------------------------------------------------------------------
extern "C" void kernel_launch(void* const* d_in, const int* in_sizes, int n_in,
                              void* d_out, int out_size, void* d_ws, size_t ws_size,
                              hipStream_t stream) {
    const float* h   = (const float*)d_in[0];
    const int*   adj = (const int*)  d_in[1];
    const float* Wq  = (const float*)d_in[2];
    const float* bq  = (const float*)d_in[3];
    const float* Wk  = (const float*)d_in[4];
    const float* bk  = (const float*)d_in[5];
    float* out = (float*)d_out;

    // workspace partition (all bf16, ~81 MB total)
    char* ws = (char*)d_ws;
    bf16_t* hbf  = (bf16_t*)ws;  ws += (size_t)MTOT * DMODEL * sizeof(bf16_t);            // 6.0 MB
    bf16_t* wqbf = (bf16_t*)ws;  ws += (size_t)EDGES * DMODEL * DMODEL * sizeof(bf16_t);  // 5.6 MB
    bf16_t* wkbf = (bf16_t*)ws;  ws += (size_t)EDGES * DMODEL * DMODEL * sizeof(bf16_t);  // 5.6 MB
    bf16_t* qbf  = (bf16_t*)ws;  ws += (size_t)EDGES * MTOT * DMODEL * sizeof(bf16_t);    // 30 MB
    bf16_t* kbf  = (bf16_t*)ws;  ws += (size_t)EDGES * MTOT * DMODEL * sizeof(bf16_t);    // 30 MB

    // 1) precision downcast
    {
        int n4 = (MTOT * DMODEL) / 4;
        cvt_f32_bf16<<<(n4 + 255) / 256, 256, 0, stream>>>(h, hbf, n4);
        n4 = (EDGES * DMODEL * DMODEL) / 4;
        cvt_f32_bf16<<<(n4 + 255) / 256, 256, 0, stream>>>(Wq, wqbf, n4);
        cvt_f32_bf16<<<(n4 + 255) / 256, 256, 0, stream>>>(Wk, wkbf, n4);
    }

    // 2) Q/K projections: 10 GEMMs of 4096x768x768 on the WMMA pipe
    {
        dim3 grid(MTOT / 16, DMODEL / 128, 2 * EDGES);
        gemm_qk_kernel<<<grid, 256, 0, stream>>>(hbf, wqbf, wkbf, bq, bk, qbf, kbf);
    }

    // 3) edge-selected attention scores straight into d_out ((H,B,N,N) layout)
    {
        dim3 grid(NNODES / 16, NHEADS, BATCH);
        score_kernel<<<grid, 256, 0, stream>>>(qbf, kbf, adj, out);
    }

    // 4) fused row softmax in place
    softmax_kernel<<<NHEADS * BATCH * NNODES, 256, 0, stream>>>(out);
}
